// WindowAttention_1D_45414984188000
// MI455X (gfx1250) — matrix operand order repaired and checked by reading, not compiled, not run on hardware
//
#include <hip/hip_runtime.h>
#include <hip/hip_bf16.h>

typedef __attribute__((ext_vector_type(16))) _Float16 v16h;
typedef __attribute__((ext_vector_type(8)))  _Float16 v8h;
typedef __attribute__((ext_vector_type(8)))  float    v8f;

#define DIMC 256
#define WSN  64
#define NH   8
#define DH   32
#define NWM  64
#define BWINB 4096
#define QKV_HALVES 67108864ull  // BWINB*NH*WSN*DH

// ---------- fragment helpers (contiguous-k order shared by A and B, wave32) ----------

__device__ __forceinline__ v8f zerov8() {
  v8f z;
#pragma unroll
  for (int i = 0; i < 8; ++i) z[i] = 0.f;
  return z;
}

// A matrix 16x32 f16: lane&15 = M row; K contiguous c..c+15 (c = kbase + 16*(lane>=16)).
__device__ __forceinline__ v16h frag_a_lds(const _Float16* tile, int stride, int kbase, int lane) {
  const _Float16* row = tile + (lane & 15) * stride;
  int c = kbase + ((lane >> 4) << 4);
  v8h lo = *(const v8h*)(row + c);
  v8h hi = *(const v8h*)(row + c + 8);
  v16h r;
#pragma unroll
  for (int i = 0; i < 8; ++i) { r[i] = lo[i]; r[i + 8] = hi[i]; }
  return r;
}

// B matrix 32x16 f16, stored transposed in LDS as mat[n][k]:
// lane&15 = N, lanes 0-15 K=kbase..+15, lanes 16-31 K=kbase+16..+31, contiguous.
__device__ __forceinline__ v16h frag_b_lds(const _Float16* mat, int stride, int nbase, int kbase, int lane) {
  const _Float16* row = mat + (nbase + (lane & 15)) * stride + kbase + ((lane >> 4) << 4);
  v8h lo = *(const v8h*)(row);
  v8h hi = *(const v8h*)(row + 8);
  v16h r;
#pragma unroll
  for (int i = 0; i < 8; ++i) { r[i] = lo[i]; r[i + 8] = hi[i]; }
  return r;
}

// B fragment straight from a row-major f32 weight (w[n][k]); weights stay L2-resident.
__device__ __forceinline__ v16h frag_b_gf32(const float* __restrict__ w, int ld, int nglob, int kglob, int lane) {
  const float* p = w + (size_t)(nglob + (lane & 15)) * ld + kglob + ((lane >> 4) << 4);
  v16h r;
#pragma unroll
  for (int i = 0; i < 16; ++i) r[i] = (_Float16)p[i];
  return r;
}

// ---------- kernel 0: fused qkv bias table (768 f32) ----------
__global__ __launch_bounds__(256) void k_bias(const float* __restrict__ q_bias,
                                              const float* __restrict__ v_bias,
                                              float* __restrict__ qkv_bias) {
  int o = blockIdx.x * 256 + threadIdx.x;  // grid 3
  float v = 0.f;
  if (o < 256) v = q_bias[o];
  else if (o >= 512) v = v_bias[o - 512];
  qkv_bias[o] = v;
}

// ---------- kernel 1: CPB MLP -> bias_table (127 x 8) ----------
__global__ __launch_bounds__(256) void k_cpb(const float* __restrict__ rel_table,
                                             const float* __restrict__ cpb_w1,
                                             const float* __restrict__ cpb_b1,
                                             const float* __restrict__ cpb_w2,
                                             float* __restrict__ bias_table) {
  __shared__ float red[256];
  int i = blockIdx.x;  // 0..126
  int tid = threadIdx.x;
  int h = tid & 7, part = tid >> 3;  // 32 partial sums per head
  float rt = rel_table[i];
  float acc = 0.f;
#pragma unroll
  for (int jj = 0; jj < 16; ++jj) {
    int j = part * 16 + jj;
    float hid = fmaxf(rt * cpb_w1[j] + cpb_b1[j], 0.f);
    acc += hid * cpb_w2[h * 512 + j];
  }
  red[tid] = acc;
  __syncthreads();
  if (tid < 8) {
    float s = 0.f;
#pragma unroll
    for (int p = 0; p < 32; ++p) s += red[tid + 8 * p];
    bias_table[i * 8 + tid] = s;
  }
}

// ---------- kernel 2: gather + 16*sigmoid -> rpb (H,64,64) f32 ----------
__global__ __launch_bounds__(256) void k_rpb(const float* __restrict__ bias_table,
                                             const int* __restrict__ rel_index,
                                             float* __restrict__ rpb) {
  int idx = blockIdx.x * 256 + threadIdx.x;  // 8*64*64 = 32768
  int h = idx >> 12, rem = idx & 4095;
  int t = rel_index[rem];
  float v = bias_table[t * 8 + h];
  rpb[idx] = 16.f / (1.f + __expf(-v));
}

// ---------- kernel 3: QKV projection (per-window 64x256 @ 256x768) ----------
__global__ __launch_bounds__(256) void k_qkv(const float* __restrict__ x,
                                             const float* __restrict__ qkv_w,
                                             const float* __restrict__ qkv_bias,
                                             _Float16* __restrict__ qkv0) {
  __shared__ __align__(16) _Float16 As[64 * 264];  // x tile f16, padded stride
  __shared__ __align__(16) _Float16 Os[64 * 776];  // staged qkv output tile
  int b = blockIdx.x;
  int tid = threadIdx.x;
  const float* xb = x + (size_t)b * (WSN * DIMC);
  for (int e = tid; e < WSN * DIMC; e += 256) {
    int r = e >> 8, c = e & 255;
    As[r * 264 + c] = (_Float16)xb[e];
  }
  __syncthreads();
  int wid = tid >> 5, lane = tid & 31;
  int mrow = (lane >> 4) << 3;  // +0 / +8 for C/D rows
  int ncol = lane & 15;
  for (int ntl = 0; ntl < 6; ++ntl) {  // 48 n-tiles / 8 waves
    int nt = wid * 6 + ntl;
    int obase = nt * 16;
    v8f acc[4];
#pragma unroll
    for (int mt = 0; mt < 4; ++mt) acc[mt] = zerov8();
    for (int kk = 0; kk < 8; ++kk) {
      int kbase = kk * 32;
      v16h bfrag = frag_b_gf32(qkv_w, DIMC, obase, kbase, lane);
#pragma unroll
      for (int mt = 0; mt < 4; ++mt) {
        v16h afrag = frag_a_lds(As + mt * 16 * 264, 264, kbase, lane);
        acc[mt] = __builtin_amdgcn_wmma_f32_16x16x32_f16(
            false, afrag, false, bfrag, (short)0, acc[mt], false, false);
      }
    }
    int o = obase + ncol;
    float bias = qkv_bias[o];  // fused table: no divergent select
#pragma unroll
    for (int mt = 0; mt < 4; ++mt) {
#pragma unroll
      for (int r = 0; r < 8; ++r) {
        int m = mt * 16 + r + mrow;
        Os[m * 776 + o] = (_Float16)(acc[mt][r] + bias);
      }
    }
  }
  __syncthreads();
  // Coalesced writeout: per (region, head) the destination is a contiguous
  // 2048-half run at qkv0 + region*QKV_HALVES + ((b*8+h)*64)*32.
  for (int idx = tid; idx < 6144; idx += 256) {  // 24 slices * 256 v8h chunks
    int rh = idx >> 8;        // 0..23 : region*8 + h
    int chunk = idx & 255;    // within slice
    int m = chunk >> 2;
    int dq = (chunk & 3) << 3;
    int region = rh >> 3, h = rh & 7;
    v8h val = *(const v8h*)(Os + m * 776 + region * 256 + h * 32 + dq);
    *(v8h*)(qkv0 + (size_t)region * QKV_HALVES +
            (((size_t)b * NH + h) * WSN + m) * DH + dq) = val;
  }
}

// ---------- kernel 4: attention per (window b, head h) ----------
__global__ __launch_bounds__(128) void k_attn(const _Float16* __restrict__ qbuf,
                                              const _Float16* __restrict__ kbuf,
                                              const _Float16* __restrict__ vbuf,
                                              const float* __restrict__ logit_scale,
                                              const float* __restrict__ mask,
                                              const float* __restrict__ rpb,
                                              _Float16* __restrict__ aout) {
  __shared__ __align__(16) _Float16 qs[64 * 32];
  __shared__ __align__(16) _Float16 ks[64 * 32];
  __shared__ __align__(16) _Float16 vsT[32 * 64];  // [d][tok]
  __shared__ __align__(16) _Float16 ps[64 * 72];   // softmax probs f16, padded
  __shared__ float sbuf[64 * 65];
  __shared__ float invq[64], invk[64], rsum[64];

  int b = blockIdx.x, h = blockIdx.y;
  int tid = threadIdx.x;
  size_t base = (((size_t)b * NH + h) * WSN) * DH;  // 2048 halves
  const v8h* qsrc = (const v8h*)(qbuf + base);
  const v8h* ksrc = (const v8h*)(kbuf + base);
  v8h* qd = (v8h*)qs;
  v8h* kd = (v8h*)ks;
  for (int i = tid; i < 256; i += 128) { qd[i] = qsrc[i]; kd[i] = ksrc[i]; }
  const _Float16* vsrc = vbuf + base;
  for (int e = tid; e < 2048; e += 128) {
    int tok = e >> 5, d = e & 31;
    vsT[d * 64 + tok] = vsrc[e];
  }
  __syncthreads();
  if (tid < 64) {
    float s = 0.f;
#pragma unroll
    for (int d = 0; d < 32; ++d) { float v = (float)qs[tid * 32 + d]; s += v * v; }
    invq[tid] = 1.f / fmaxf(sqrtf(s), 1e-12f);
  } else {
    int r = tid - 64;
    float s = 0.f;
#pragma unroll
    for (int d = 0; d < 32; ++d) { float v = (float)ks[r * 32 + d]; s += v * v; }
    invk[r] = 1.f / fmaxf(sqrtf(s), 1e-12f);
  }
  __syncthreads();

  float hscale = __expf(fminf(logit_scale[h], 4.6051701859880914f));  // ln(100)
  int wid = tid >> 5, lane = tid & 31;
  int mrow = (lane >> 4) << 3, ncol = lane & 15;
  const float* maskw = mask + (size_t)(b & (NWM - 1)) * 4096;
  const float* rpbh = rpb + (size_t)h * 4096;

  // GEMM1: s = q @ k^T  (M=64 split over 4 waves, K=32 single step)
  v16h aq = frag_a_lds(qs + wid * 16 * 32, 32, 0, lane);
  for (int nt = 0; nt < 4; ++nt) {
    v16h bk = frag_b_lds(ks, 32, nt * 16, 0, lane);
    v8f acc = zerov8();
    acc = __builtin_amdgcn_wmma_f32_16x16x32_f16(false, aq, false, bk, (short)0, acc, false, false);
#pragma unroll
    for (int r = 0; r < 8; ++r) {
      int m = wid * 16 + r + mrow;
      int n = nt * 16 + ncol;
      sbuf[m * 65 + n] = acc[r] * invq[m] * invk[n] * hscale + rpbh[m * 64 + n] + maskw[m * 64 + n];
    }
  }
  __syncthreads();

  // softmax rows
  if (tid < 64) {
    float mx = -3.4e38f;
    for (int j = 0; j < 64; ++j) mx = fmaxf(mx, sbuf[tid * 65 + j]);
    float sum = 0.f;
    for (int j = 0; j < 64; ++j) {
      float e = __expf(sbuf[tid * 65 + j] - mx);
      sum += e;
      ps[tid * 72 + j] = (_Float16)e;
    }
    rsum[tid] = 1.f / sum;
  }
  __syncthreads();
  for (int e = tid; e < 4096; e += 128) {
    int m = e >> 6, j = e & 63;
    ps[m * 72 + j] = (_Float16)((float)ps[m * 72 + j] * rsum[m]);
  }
  __syncthreads();

  // GEMM2: out = p (64x64) @ v (64x32)
  v16h pa0 = frag_a_lds(ps + wid * 16 * 72, 72, 0, lane);
  v16h pa1 = frag_a_lds(ps + wid * 16 * 72, 72, 32, lane);
  _Float16* od = aout + ((size_t)b * 64) * 256 + h * 32;
  for (int nt = 0; nt < 2; ++nt) {
    v8f acc = zerov8();
    v16h bv0 = frag_b_lds(vsT, 64, nt * 16, 0, lane);
    acc = __builtin_amdgcn_wmma_f32_16x16x32_f16(false, pa0, false, bv0, (short)0, acc, false, false);
    v16h bv1 = frag_b_lds(vsT, 64, nt * 16, 32, lane);
    acc = __builtin_amdgcn_wmma_f32_16x16x32_f16(false, pa1, false, bv1, (short)0, acc, false, false);
#pragma unroll
    for (int r = 0; r < 8; ++r) {
      int m = wid * 16 + r + mrow;
      int d = nt * 16 + ncol;
      od[(size_t)m * 256 + d] = (_Float16)acc[r];
    }
  }
}

// ---------- kernel 5: output projection (64x256 @ 256x256) + bias, f32 out ----------
__global__ __launch_bounds__(256) void k_proj(const _Float16* __restrict__ aout,
                                              const float* __restrict__ proj_w,
                                              const float* __restrict__ proj_b,
                                              float* __restrict__ out) {
  __shared__ __align__(16) _Float16 As[64 * 264];
  int b = blockIdx.x, tid = threadIdx.x;
  const v8h* src = (const v8h*)(aout + (size_t)b * (64 * 256));
  for (int i = tid; i < 2048; i += 256) {
    int r = i >> 5, c = i & 31;  // 32 v8h chunks per 256-half row
    *(v8h*)(As + r * 264 + c * 8) = src[i];
  }
  __syncthreads();
  int wid = tid >> 5, lane = tid & 31;
  int mrow = (lane >> 4) << 3, ncol = lane & 15;
  float* ob = out + (size_t)b * (64 * 256);
  for (int ntl = 0; ntl < 2; ++ntl) {  // 16 n-tiles / 8 waves
    int nt = wid * 2 + ntl;
    int obase = nt * 16;
    v8f acc[4];
#pragma unroll
    for (int mt = 0; mt < 4; ++mt) acc[mt] = zerov8();
    for (int kk = 0; kk < 8; ++kk) {
      v16h bfrag = frag_b_gf32(proj_w, DIMC, obase, kk * 32, lane);
#pragma unroll
      for (int mt = 0; mt < 4; ++mt) {
        v16h afrag = frag_a_lds(As + mt * 16 * 264, 264, kk * 32, lane);
        acc[mt] = __builtin_amdgcn_wmma_f32_16x16x32_f16(
            false, afrag, false, bfrag, (short)0, acc[mt], false, false);
      }
    }
    int o = obase + ncol;
    float pb = proj_b[o];
#pragma unroll
    for (int mt = 0; mt < 4; ++mt) {
#pragma unroll
      for (int r = 0; r < 8; ++r) {
        int m = mt * 16 + r + mrow;
        ob[(size_t)m * 256 + o] = acc[mt][r] + pb;
      }
    }
  }
}

extern "C" void kernel_launch(void* const* d_in, const int* in_sizes, int n_in,
                              void* d_out, int out_size, void* d_ws, size_t ws_size,
                              hipStream_t stream) {
  (void)in_sizes; (void)n_in; (void)out_size; (void)ws_size;
  const float* x           = (const float*)d_in[0];
  const float* mask        = (const float*)d_in[1];
  const float* qkv_w       = (const float*)d_in[2];
  const float* q_bias      = (const float*)d_in[3];
  const float* v_bias      = (const float*)d_in[4];
  const float* logit_scale = (const float*)d_in[5];
  const float* cpb_w1      = (const float*)d_in[6];
  const float* cpb_b1      = (const float*)d_in[7];
  const float* cpb_w2      = (const float*)d_in[8];
  const float* proj_w      = (const float*)d_in[9];
  const float* proj_b      = (const float*)d_in[10];
  const float* rel_table   = (const float*)d_in[11];
  const int*   rel_index   = (const int*)d_in[12];
  float* out = (float*)d_out;

  char* ws = (char*)d_ws;
  float* bias_table = (float*)(ws);                   // 127*8 f32
  float* rpb        = (float*)(ws + 4096);            // 8*64*64 f32
  float* qkv_bias   = (float*)(ws + 256 * 1024);      // 768 f32
  _Float16* qbuf = (_Float16*)(ws + (1ull << 20));    // contiguous q|k|v|aout
  _Float16* kbuf = qbuf + QKV_HALVES;
  _Float16* vbuf = kbuf + QKV_HALVES;
  _Float16* abuf = vbuf + QKV_HALVES;                 // (B_, N, C) f16

  k_bias<<<3, 256, 0, stream>>>(q_bias, v_bias, qkv_bias);
  k_cpb<<<127, 256, 0, stream>>>(rel_table, cpb_w1, cpb_b1, cpb_w2, bias_table);
  k_rpb<<<128, 256, 0, stream>>>(bias_table, rel_index, rpb);
  k_qkv<<<BWINB, 256, 0, stream>>>(x, qkv_w, qkv_bias, qbuf);
  dim3 ga(BWINB, NH);
  k_attn<<<ga, 128, 0, stream>>>(qbuf, kbuf, vbuf, logit_scale, mask, rpb, abuf);
  k_proj<<<BWINB, 256, 0, stream>>>(abuf, proj_w, proj_b, out);
}
